// TransitionUp_20890720928296
// MI455X (gfx1250) — matrix operand order repaired and checked by reading, not compiled, not run on hardware
//
#include <hip/hip_runtime.h>
#include <hip/hip_bf16.h>
#include <stdint.h>

typedef __attribute__((ext_vector_type(16))) __bf16         v16bf;
typedef __attribute__((ext_vector_type(8)))  float          v8f;
typedef __attribute__((ext_vector_type(16))) unsigned short u16x16;
typedef __attribute__((ext_vector_type(8)))  unsigned short u16x8;
typedef __attribute__((ext_vector_type(4)))  unsigned short u16x4;
typedef __attribute__((ext_vector_type(4)))  float          f32x4;

#define RSPLIT 32
#define TM 128
#define TN 128
#define TK 32
#define LDP 40   // padded LDS row stride (ushorts): 80B => 20-bank stride, conflict-free

__device__ __forceinline__ unsigned short f2bf(float f) {
  union { float f; unsigned u; } a; a.f = f;
  unsigned u = a.u + 0x7FFFu + ((a.u >> 16) & 1u);   // round-to-nearest-even
  return (unsigned short)(u >> 16);
}

// Async copy 16B/lane from global to LDS (CDNA5 path, tracked by ASYNCcnt).
__device__ __forceinline__ void async_g2l_b128(unsigned lds_off, const void* gaddr) {
  asm volatile("global_load_async_to_lds_b128 %0, %1, off"
               :: "v"(lds_off), "v"(gaddr) : "memory");
}
__device__ __forceinline__ void wait_asynccnt0() {
  asm volatile("s_wait_asynccnt 0x0" ::: "memory");
}

// ---------------- pack W1a = W1[:, :C] into bf16, row-major [n][k] ----------------
__global__ void k_pack_w1a(const float* __restrict__ W1, unsigned short* __restrict__ Bp, int C) {
  int g4 = (blockIdx.x * blockDim.x + threadIdx.x) * 4;
  if (g4 >= C * C) return;
  int n = g4 / C, k = g4 % C;
  f32x4 w = *(const f32x4*)(W1 + (size_t)n * (2 * C) + k);
  u16x4 p = { f2bf(w[0]), f2bf(w[1]), f2bf(w[2]), f2bf(w[3]) };
  *(u16x4*)(Bp + g4) = p;
}

// ---------------- per-segment partial column sums (deterministic split) ----------------
__global__ void k_segpart(const float* __restrict__ x, const int* __restrict__ o,
                          float* __restrict__ part, int C, int B) {
  int b   = blockIdx.z;
  int col = blockIdx.y * 128 + threadIdx.x;
  int start = (b == 0) ? 0 : o[b - 1];
  int end   = o[b];
  int count = end - start;
  int chunk = (count + RSPLIT - 1) / RSPLIT;
  int r0 = start + blockIdx.x * chunk;
  int r1 = r0 + chunk; if (r1 > end) r1 = end;
  float acc = 0.f;
  for (int m = r0; m < r1; ++m) acc += x[(size_t)m * C + col];
  part[((size_t)blockIdx.x * B + b) * C + col] = acc;
}

// ---- means -> pooled=relu(means@W2^T+b2) -> segbias = pooled@W1b^T + b1 (per segment) ----
__global__ void k_seg_finalize(const float* __restrict__ part, const int* __restrict__ o,
                               const float* __restrict__ W2, const float* __restrict__ b2,
                               const float* __restrict__ W1, const float* __restrict__ b1,
                               float* __restrict__ segbias, int C, int B) {
  __shared__ float means_s[512];
  __shared__ float pooled_s[512];
  int b = blockIdx.x;
  int j = threadIdx.x;
  int start = (b == 0) ? 0 : o[b - 1];
  float inv_cnt = 1.0f / (float)(o[b] - start);
  float s = 0.f;
  for (int r = 0; r < RSPLIT; ++r) s += part[((size_t)r * B + b) * C + j];
  means_s[j] = s * inv_cnt;
  __syncthreads();
  const float* w2r = W2 + (size_t)j * C;
  float p = b2[j];
  for (int k = 0; k < C; k += 4) {
    f32x4 m4 = *(const f32x4*)(means_s + k);
    f32x4 w4 = *(const f32x4*)(w2r + k);
    p += m4[0]*w4[0] + m4[1]*w4[1] + m4[2]*w4[2] + m4[3]*w4[3];
  }
  pooled_s[j] = fmaxf(p, 0.f);
  __syncthreads();
  const float* w1r = W1 + (size_t)j * (2 * C) + C;   // row j of W1b
  float sb = b1[j];
  for (int k = 0; k < C; k += 4) {
    f32x4 p4 = *(const f32x4*)(pooled_s + k);
    f32x4 w4 = *(const f32x4*)(w1r + k);
    sb += p4[0]*w4[0] + p4[1]*w4[1] + p4[2]*w4[2] + p4[3]*w4[3];
  }
  segbias[(size_t)b * C + j] = sb;
}

// ---------------- main GEMM: h = bf16(x) @ bf16(W1a)^T + segbias[seg], + BN partials -----
__global__ __launch_bounds__(256)
void k_gemm(const float* __restrict__ x, const unsigned short* __restrict__ Bp,
            const float* __restrict__ segbias, const int* __restrict__ o, int B,
            float* __restrict__ out, float* __restrict__ colS, float* __restrict__ colQ,
            int C) {
  __shared__ unsigned short As[2][TM][LDP];     // double-buffered
  __shared__ unsigned short Bs[2][TN][LDP];
  __shared__ float redS[4][TN];
  __shared__ float redQ[4][TN];
  __shared__ int o_s[16];
  __shared__ int segrow_s[TM];

  const int t    = threadIdx.x;
  const int lane = t & 31;
  const int w    = t >> 5;
  const int wr   = w >> 1;          // 0..3 : 32-row slab
  const int wc   = w & 1;           // 0..1 : 64-col slab
  const int m0   = blockIdx.x * TM;
  const int n0   = blockIdx.y * TN;
  if (t < B) o_s[t] = o[t];

  v8f acc[2][4];
  #pragma unroll
  for (int mi = 0; mi < 2; ++mi)
    #pragma unroll
    for (int ni = 0; ni < 4; ++ni)
      #pragma unroll
      for (int g = 0; g < 8; ++g) acc[mi][ni][g] = 0.f;

  const int lrow = t >> 1;          // 0..127 : tile row loaded by this thread (A path)
  const int lk   = (t & 1) * 16;    // 0 / 16 : k sub-chunk (A path)
  const int r    = lane & 15;
  const int hi   = lane >> 4;
  // B-tile async-copy mapping: 512 x 16B chunks per tile, 2 per thread
  const int brow0 = t >> 2;                  // rows 0..63
  const int bko0  = (t & 3) * 8;             // bf16 offset within row
  const int brow1 = (t + 256) >> 2;          // rows 64..127

  auto stage = [&](int k0, int sel) {
    // B tile: pure copy, async global->LDS (ASYNCcnt), no VGPR roundtrip
    async_g2l_b128((unsigned)(uintptr_t)&Bs[sel][brow0][bko0],
                   Bp + (size_t)(n0 + brow0) * C + k0 + bko0);
    async_g2l_b128((unsigned)(uintptr_t)&Bs[sel][brow1][bko0],
                   Bp + (size_t)(n0 + brow1) * C + k0 + bko0);
    // A tile: fp32 -> bf16 through VGPRs
    const float* xp = x + (size_t)(m0 + lrow) * C + k0 + lk;
    f32x4 a0 = *(const f32x4*)(xp + 0);
    f32x4 a1 = *(const f32x4*)(xp + 4);
    f32x4 a2 = *(const f32x4*)(xp + 8);
    f32x4 a3 = *(const f32x4*)(xp + 12);
    u16x8 pa0 = { f2bf(a0[0]), f2bf(a0[1]), f2bf(a0[2]), f2bf(a0[3]),
                  f2bf(a1[0]), f2bf(a1[1]), f2bf(a1[2]), f2bf(a1[3]) };
    u16x8 pa1 = { f2bf(a2[0]), f2bf(a2[1]), f2bf(a2[2]), f2bf(a2[3]),
                  f2bf(a3[0]), f2bf(a3[1]), f2bf(a3[2]), f2bf(a3[3]) };
    *(u16x8*)&As[sel][lrow][lk]     = pa0;
    *(u16x8*)&As[sel][lrow][lk + 8] = pa1;
  };

  const int nsteps = C / TK;
  stage(0, 0);
  wait_asynccnt0();
  __syncthreads();

  for (int i = 0; i < nsteps; ++i) {
    const int sel = i & 1;
    if (i + 1 < nsteps) stage((i + 1) * TK, sel ^ 1);   // overlap with compute below

    // ---- fragments per ISA 16-bit layouts ----
    v16bf af[2], bfm[4];
    #pragma unroll
    for (int mi = 0; mi < 2; ++mi) {               // A: lanes<16 K{0-7,16-23}; lanes>=16 K{8-15,24-31}
      const unsigned short* p = &As[sel][wr * 32 + mi * 16 + r][hi * 8];
      u16x8 lo = *(const u16x8*)p;
      u16x8 h8 = *(const u16x8*)(p + 16);
      u16x16 c;
      #pragma unroll
      for (int j = 0; j < 8; ++j) { c[j] = lo[j]; c[8 + j] = h8[j]; }
      af[mi] = __builtin_bit_cast(v16bf, c);
    }
    #pragma unroll
    for (int ni = 0; ni < 4; ++ni) {               // B: lanes<16 K{0-7,16-23}; lanes>=16 K{8-15,24-31} (N = lane%16)
      const unsigned short* p = &Bs[sel][wc * 64 + ni * 16 + r][hi * 8];
      u16x8 lo = *(const u16x8*)p;
      u16x8 h8 = *(const u16x8*)(p + 16);
      u16x16 c;
      #pragma unroll
      for (int j = 0; j < 8; ++j) { c[j] = lo[j]; c[8 + j] = h8[j]; }
      bfm[ni] = __builtin_bit_cast(v16bf, c);
    }
    #pragma unroll
    for (int mi = 0; mi < 2; ++mi)
      #pragma unroll
      for (int ni = 0; ni < 4; ++ni)
        acc[mi][ni] = __builtin_amdgcn_wmma_f32_16x16x32_bf16(
            false, af[mi], false, bfm[ni], (short)0, acc[mi][ni], false, false);

    wait_asynccnt0();               // next tile's B landed
    __syncthreads();                // next tile's A stores + this tile's frag reads done
  }

  // ---- per-row segment ids: branchless sorted-offset count ----
  if (t < TM) {
    int m = m0 + t;
    int s = 0;
    for (int i = 0; i < B - 1; ++i) s += (m >= o_s[i]) ? 1 : 0;
    segrow_s[t] = s;
  }
  __syncthreads();

  const int lcol = lane & 15;
  const int lhi  = lane >> 4;
  const bool uniform = (segrow_s[0] == segrow_s[TM - 1]);  // block-uniform fast path
  const int  s0 = segrow_s[0];
  int segid[2][8];
  #pragma unroll
  for (int mi = 0; mi < 2; ++mi)
    #pragma unroll
    for (int g = 0; g < 8; ++g)
      segid[mi][g] = segrow_s[wr * 32 + mi * 16 + lhi * 8 + g];

  #pragma unroll
  for (int ni = 0; ni < 4; ++ni) {
    const int nglob = n0 + wc * 64 + ni * 16 + lcol;
    float s1 = 0.f, s2 = 0.f;
    if (uniform) {
      const float bias = segbias[(size_t)s0 * C + nglob];
      #pragma unroll
      for (int mi = 0; mi < 2; ++mi) {
        #pragma unroll
        for (int g = 0; g < 8; ++g) {
          int m = m0 + wr * 32 + mi * 16 + lhi * 8 + g;  // C/D layout: lanes>=16 hold M=8+g
          float h = acc[mi][ni][g] + bias;
          out[(size_t)m * C + nglob] = h;
          s1 += h; s2 += h * h;
        }
      }
    } else {
      #pragma unroll
      for (int mi = 0; mi < 2; ++mi) {
        #pragma unroll
        for (int g = 0; g < 8; ++g) {
          int m = m0 + wr * 32 + mi * 16 + lhi * 8 + g;
          float h = acc[mi][ni][g] + segbias[(size_t)segid[mi][g] * C + nglob];
          out[(size_t)m * C + nglob] = h;
          s1 += h; s2 += h * h;
        }
      }
    }
    s1 += __shfl_xor(s1, 16);
    s2 += __shfl_xor(s2, 16);
    if (lhi == 0) {
      redS[wr][wc * 64 + ni * 16 + lcol] = s1;
      redQ[wr][wc * 64 + ni * 16 + lcol] = s2;
    }
  }
  __syncthreads();
  if (t < TN) {
    float s = redS[0][t] + redS[1][t] + redS[2][t] + redS[3][t];
    float q = redQ[0][t] + redQ[1][t] + redQ[2][t] + redQ[3][t];
    colS[(size_t)blockIdx.x * C + n0 + t] = s;
    colQ[(size_t)blockIdx.x * C + n0 + t] = q;
  }
}

// ---------------- fold per-M-tile partials into BN scale/shift ----------------
__global__ void k_bn_reduce(const float* __restrict__ colS, const float* __restrict__ colQ,
                            const float* __restrict__ gamma, const float* __restrict__ beta,
                            float* __restrict__ scale, float* __restrict__ shift,
                            int C, int MT, float invN) {
  int c = blockIdx.x * blockDim.x + threadIdx.x;
  if (c >= C) return;
  float s = 0.f, q = 0.f;
  for (int i = 0; i < MT; ++i) { s += colS[(size_t)i * C + c]; q += colQ[(size_t)i * C + c]; }
  float mu  = s * invN;
  float var = q * invN - mu * mu;
  float sc  = gamma[c] * rsqrtf(var + 1e-5f);
  scale[c] = sc;
  shift[c] = beta[c] - mu * sc;
}

// ---------------- in-place BN apply + ReLU ----------------
__global__ void k_finalize(float* __restrict__ out, const float* __restrict__ scale,
                           const float* __restrict__ shift, int C, size_t total4) {
  size_t i = blockIdx.x * (size_t)blockDim.x + threadIdx.x;
  if (i >= total4) return;
  size_t e0 = i * 4;
  int c = (int)(e0 % (size_t)C);
  f32x4 h  = *(f32x4*)(out + e0);
  f32x4 sc = *(const f32x4*)(scale + c);
  f32x4 sh = *(const f32x4*)(shift + c);
  f32x4 r;
  #pragma unroll
  for (int j = 0; j < 4; ++j) r[j] = fmaxf(h[j] * sc[j] + sh[j], 0.f);
  *(f32x4*)(out + e0) = r;
}

extern "C" void kernel_launch(void* const* d_in, const int* in_sizes, int n_in,
                              void* d_out, int out_size, void* d_ws, size_t ws_size,
                              hipStream_t stream) {
  const float* x     = (const float*)d_in[0];
  const int*   o     = (const int*)d_in[1];
  const float* W1    = (const float*)d_in[2];
  const float* b1    = (const float*)d_in[3];
  const float* gamma = (const float*)d_in[4];
  const float* beta  = (const float*)d_in[5];
  const float* W2    = (const float*)d_in[6];
  const float* b2    = (const float*)d_in[7];
  float* out = (float*)d_out;

  const int  C  = in_sizes[3];                    // 512
  const int  B  = in_sizes[1];                    // 8
  const long long N = (long long)in_sizes[0] / C; // 131072
  const int  MT = (int)(N / TM);                  // 1024 M-tiles

  char* ws = (char*)d_ws;
  size_t off = 0;
  auto alloc = [&](size_t bytes) -> void* {
    void* p = ws + off;
    off += (bytes + 255) & ~(size_t)255;
    return p;
  };
  unsigned short* Bpack  = (unsigned short*)alloc((size_t)C * C * 2);
  float* segpart = (float*)alloc((size_t)RSPLIT * B * C * 4);
  float* segbias = (float*)alloc((size_t)B * C * 4);
  float* colS    = (float*)alloc((size_t)MT * C * 4);
  float* colQ    = (float*)alloc((size_t)MT * C * 4);
  float* scale   = (float*)alloc((size_t)C * 4);
  float* shift   = (float*)alloc((size_t)C * 4);

  k_pack_w1a<<<(C * C / 4 + 255) / 256, 256, 0, stream>>>(W1, Bpack, C);

  dim3 gs(RSPLIT, C / 128, B);
  k_segpart<<<gs, 128, 0, stream>>>(x, o, segpart, C, B);

  k_seg_finalize<<<B, C, 0, stream>>>(segpart, o, W2, b2, W1, b1, segbias, C, B);

  dim3 gg(MT, C / TN);
  k_gemm<<<gg, 256, 0, stream>>>(x, Bpack, segbias, o, B, out, colS, colQ, C);

  k_bn_reduce<<<(C + 255) / 256, 256, 0, stream>>>(colS, colQ, gamma, beta, scale, shift,
                                                   C, MT, 1.0f / (float)N);

  size_t total4 = (size_t)N * C / 4;
  k_finalize<<<(unsigned)((total4 + 255) / 256), 256, 0, stream>>>(out, scale, shift, C, total4);
}
